// AttentionGRU_64871186038840
// MI455X (gfx1250) — matrix operand order repaired and checked by reading, not compiled, not run on hardware
//
#include <hip/hip_runtime.h>
#include <hip/hip_bf16.h>
#include <math.h>

// Problem dims (from reference): B=512, T_DEC=64, T_ENC=128, E=128, H=256
#define B_   512
#define TDEC 64
#define TENC 128
#define E_   128
#define H_   256
#define H2_  512

typedef __attribute__((ext_vector_type(16))) __bf16 v16bf;
typedef __attribute__((ext_vector_type(8)))  float  v8f;

union ABf { unsigned int u[8]; v16bf v; };

__device__ __forceinline__ unsigned short f32_to_bf16(float f) {
  unsigned int x = __float_as_uint(f);
  unsigned int r = (x + 0x7FFFu + ((x >> 16) & 1u)) >> 16;  // round-to-nearest-even
  return (unsigned short)r;
}
__device__ __forceinline__ float sigmoidf_(float x) { return 1.f / (1.f + __expf(-x)); }

__device__ __forceinline__ v8f wmma_bf16(const ABf& a, const ABf& b, v8f c) {
  // D = A(16x32 bf16) x B(32x16 bf16) + C(16x16 f32)
  return __builtin_amdgcn_wmma_f32_16x16x32_bf16(false, a.v, false, b.v, (short)0, c,
                                                 false, false);
}

// ---------------------------------------------------------------- f32 -> bf16
__global__ __launch_bounds__(256) void k_cvt_bf16(const float* __restrict__ src,
                                                  unsigned short* __restrict__ dst, int n) {
  int i = blockIdx.x * blockDim.x + threadIdx.x;
  if (i < n) dst[i] = f32_to_bf16(src[i]);
}

// -------------------------------------------------- fold enc_projection into u, c0
// u[j] = sum_h Wenc[h, j] * v[h]    (Wenc: [H, 2H]),   c0 = benc . v
__global__ __launch_bounds__(512) void k_u_c0(const float* __restrict__ wenc,
                                              const float* __restrict__ benc,
                                              const float* __restrict__ vE,
                                              float* __restrict__ u, float* __restrict__ c0) {
  int j = threadIdx.x;
  if (j < H2_) {
    float acc = 0.f;
    for (int h = 0; h < H_; ++h) acc += wenc[h * H2_ + j] * vE[h];
    u[j] = acc;
  }
  if (j == 0) {
    float a = 0.f;
    for (int h = 0; h < H_; ++h) a += benc[h] * vE[h];
    *c0 = a;
  }
}

// -------------------------------------------------- attention (state-independent)
// logit[b,j] = all_h_enc[b,j,:] . u + c0 ; mask ; softmax over j ; ct = sum_j a_j * h_j
__global__ __launch_bounds__(256) void k_attention(const float* __restrict__ henc,
                                                   const float* __restrict__ u,
                                                   const float* __restrict__ c0p,
                                                   float* __restrict__ ctF,
                                                   unsigned short* __restrict__ ctB) {
  __shared__ float uS[H2_];
  __shared__ float part[256];
  __shared__ int   zpart[256];
  __shared__ float sc[TENC];
  __shared__ float red[256];
  const int b = blockIdx.x, tid = threadIdx.x;
  const float* hb = henc + (size_t)b * TENC * H2_;
  uS[tid] = u[tid]; uS[tid + 256] = u[tid + 256];
  __syncthreads();

  const int j = tid >> 1, half = tid & 1;
  const float* row = hb + j * H2_ + half * 256;
  float acc = 0.f; int anyz = 0;
  for (int k = 0; k < 256; ++k) {
    float x = row[k];
    acc += x * uS[half * 256 + k];
    anyz |= (x == 0.f);
  }
  part[tid] = acc; zpart[tid] = anyz;
  __syncthreads();
  if (tid < TENC) {
    float s = part[2 * tid] + part[2 * tid + 1] + c0p[0];
    int az = zpart[2 * tid] | zpart[2 * tid + 1];
    sc[tid] = az ? -__builtin_inff() : s;
  }
  __syncthreads();
  // softmax over 128
  red[tid] = (tid < TENC) ? sc[tid] : -__builtin_inff();
  __syncthreads();
  for (int s2 = 128; s2 > 0; s2 >>= 1) {
    if (tid < s2) red[tid] = fmaxf(red[tid], red[tid + s2]);
    __syncthreads();
  }
  float mx = red[0];
  __syncthreads();
  float e = 0.f;
  if (tid < TENC) e = __expf(sc[tid] - mx);
  red[tid] = e;
  __syncthreads();
  for (int s2 = 128; s2 > 0; s2 >>= 1) {
    if (tid < s2) red[tid] = red[tid] + red[tid + s2];
    __syncthreads();
  }
  float inv = 1.f / red[0];
  __syncthreads();
  if (tid < TENC) sc[tid] = e * inv;
  __syncthreads();
  // context: each thread owns 2 of the 512 columns
  for (int c = 0; c < 2; ++c) {
    int k = tid + c * 256;
    float a2 = 0.f;
    for (int jj = 0; jj < TENC; ++jj) a2 += sc[jj] * hb[jj * H2_ + k];
    ctF[(size_t)b * H2_ + k] = a2;
    ctB[(size_t)b * H2_ + k] = f32_to_bf16(a2);
  }
}

// -------------------------------------------------- Cgate = ct @ Wctx_g^T + b_ctx + b_in
__global__ __launch_bounds__(128) void k_ctx_gate(const unsigned* __restrict__ ctbf,
                                                  const unsigned* __restrict__ wctxbf,
                                                  const float* __restrict__ bcr,
                                                  const float* __restrict__ bcz,
                                                  const float* __restrict__ bcc,
                                                  const float* __restrict__ bir,
                                                  const float* __restrict__ biz,
                                                  const float* __restrict__ bic,
                                                  float* __restrict__ Cgate) {
  const int wave = threadIdx.x >> 5;
  const int lane = threadIdx.x & 31, lm = lane & 15, lh = lane >> 4;
  const int mt = blockIdx.x;                // 512/16 = 32 row tiles
  const int nt = blockIdx.y * 4 + wave;     // 256/16 = 16 col tiles
  const int g  = blockIdx.z;
  const unsigned* W = wctxbf + (size_t)g * (H_ * H2_ / 2);
  v8f acc = {};
#pragma unroll 4
  for (int kb = 0; kb < 16; ++kb) {  // K = 512 = 16 * 32
    ABf a, bT;
#pragma unroll
    for (int v = 0; v < 8; ++v)
      a.u[v] = ctbf[(mt * 16 + lm) * 256 + kb * 16 + ((v >> 2) << 3) + (lh << 2) + (v & 3)];
#pragma unroll
    for (int v = 0; v < 8; ++v)
      bT.u[v] = W[(nt * 16 + lm) * 256 + kb * 16 + ((v >> 2) << 3) + (lh << 2) + (v & 3)];
    acc = wmma_bf16(a, bT, acc);
  }
  const int n = nt * 16 + lm;
  const float* bc = (g == 0) ? bcr : (g == 1) ? bcz : bcc;
  const float* bi = (g == 0) ? bir : (g == 1) ? biz : bic;
  const float bias = bc[n] + bi[n];
#pragma unroll
  for (int v = 0; v < 8; ++v) {
    const int m = mt * 16 + (lh << 3) + v;
    Cgate[(size_t)(m * 3 + g) * H_ + n] = acc[v] + bias;
  }
}

// -------------------------------------------------- X[b,t,g,:] = emb @ Win_g^T + Cgate
__global__ __launch_bounds__(128) void k_emb_gate(const unsigned* __restrict__ embbf,
                                                  const unsigned* __restrict__ winbf,
                                                  const float* __restrict__ Cgate,
                                                  float* __restrict__ X) {
  const int wave = threadIdx.x >> 5;
  const int lane = threadIdx.x & 31, lm = lane & 15, lh = lane >> 4;
  const int mt = blockIdx.x;                // (B*T)/16 = 2048 row tiles
  const int nt = blockIdx.y * 4 + wave;
  const int g  = blockIdx.z;
  const unsigned* W = winbf + (size_t)g * (H_ * E_ / 2);
  v8f acc = {};
#pragma unroll
  for (int kb = 0; kb < 4; ++kb) {  // K = 128 = 4 * 32
    ABf a, bT;
#pragma unroll
    for (int v = 0; v < 8; ++v)
      a.u[v] = embbf[(mt * 16 + lm) * 64 + kb * 16 + ((v >> 2) << 3) + (lh << 2) + (v & 3)];
#pragma unroll
    for (int v = 0; v < 8; ++v)
      bT.u[v] = W[(nt * 16 + lm) * 64 + kb * 16 + ((v >> 2) << 3) + (lh << 2) + (v & 3)];
    acc = wmma_bf16(a, bT, acc);
  }
  const int n = nt * 16 + lm;
#pragma unroll
  for (int v = 0; v < 8; ++v) {
    const int row = mt * 16 + (lh << 3) + v;
    const int bb = row >> 6;  // TDEC == 64
    X[(size_t)(row * 3 + g) * H_ + n] = acc[v] + Cgate[(size_t)(bb * 3 + g) * H_ + n];
  }
}

// -------------------------------------------------- GRU recurrence, state in LDS
// 512 threads = 16 waves; each wave owns ONE 16-column tile of all three hidden
// gate matrices, preloaded into VGPRs once (weights are t-invariant). Per step,
// only the precomputed gate inputs X (read-once stream) touch global memory.
__global__ __launch_bounds__(512) void k_recurrence(const float* __restrict__ s_prev,
                                                    const float* __restrict__ X,
                                                    const unsigned* __restrict__ whid,
                                                    const float* __restrict__ bhr,
                                                    const float* __restrict__ bhz,
                                                    const float* __restrict__ bhc,
                                                    float* __restrict__ out) {
  __shared__ float          sF[16 * 256];  // f32 master state
  __shared__ unsigned short sB[16 * 256];  // bf16 state (A operand)
  __shared__ unsigned short aB[16 * 256];  // bf16 rt*s  (A operand)
  const int b0 = blockIdx.x * 16;
  for (int i = threadIdx.x; i < 16 * 256; i += 512) {
    float v = s_prev[(size_t)b0 * 256 + i];
    sF[i] = v;
    sB[i] = f32_to_bf16(v);
  }
  __syncthreads();
  const int wave = threadIdx.x >> 5;
  const int lane = threadIdx.x & 31, lm = lane & 15, lh = lane >> 4;
  const unsigned* sBu = (const unsigned*)sB;
  const unsigned* aBu = (const unsigned*)aB;
  const unsigned* Wr = whid;
  const unsigned* Wz = whid + 32768;  // H*H/2 dwords per gate
  const unsigned* Wc = whid + 65536;
  const int n0 = wave * 16;   // this wave's column tile
  const int nb = n0 + lm;     // this lane's output column
  const float bR = bhr[nb], bZ = bhz[nb], bC = bhc[nb];

  // ---- preload this wave's weight slices into registers (held across t loop)
  ABf wr[8], wz[8], wc[8];
#pragma unroll
  for (int kb = 0; kb < 8; ++kb) {
    const int kd = kb * 16 + (lh << 2);
#pragma unroll
    for (int v = 0; v < 8; ++v) {
      wr[kb].u[v] = Wr[(n0 + lm) * 128 + kd + ((v >> 2) << 3) + (v & 3)];
      wz[kb].u[v] = Wz[(n0 + lm) * 128 + kd + ((v >> 2) << 3) + (v & 3)];
      wc[kb].u[v] = Wc[(n0 + lm) * 128 + kd + ((v >> 2) << 3) + (v & 3)];
    }
  }

  for (int t = 0; t < TDEC; ++t) {
    if (t + 1 < TDEC)
      __builtin_prefetch((const void*)(X + (size_t)((b0 * TDEC + t + 1) * 3) * 256), 0, 1);

    // ---- GEMM1: s @ Whr^T, s @ Whz^T  (A from LDS, B from registers)
    v8f accR = {}, accZ = {};
#pragma unroll
    for (int kb = 0; kb < 8; ++kb) {  // K = 256 = 8 * 32
      ABf a;
#pragma unroll
      for (int v = 0; v < 8; ++v)
        a.u[v] = sBu[lm * 128 + kb * 16 + ((v >> 2) << 3) + (lh << 2) + (v & 3)];
      accR = wmma_bf16(a, wr[kb], accR);
      accZ = wmma_bf16(a, wz[kb], accZ);
    }
    // ---- gates r, z;  build A2 = bf16(rt * s)
    float zt[8];
#pragma unroll
    for (int v = 0; v < 8; ++v) {
      const int m = (lh << 3) + v;
      const int bb = b0 + m;
      const float* Xrow = X + (size_t)((bb * TDEC + t) * 3) * 256;
      float r = sigmoidf_(accR[v] + Xrow[nb] + bR);
      float z = sigmoidf_(accZ[v] + Xrow[256 + nb] + bZ);
      zt[v] = z;
      aB[m * 256 + nb] = f32_to_bf16(r * sF[m * 256 + nb]);
    }
    __syncthreads();
    // ---- GEMM2: (rt*s) @ Whc^T
    v8f accC = {};
#pragma unroll
    for (int kb = 0; kb < 8; ++kb) {
      ABf a;
#pragma unroll
      for (int v = 0; v < 8; ++v)
        a.u[v] = aBu[lm * 128 + kb * 16 + ((v >> 2) << 3) + (lh << 2) + (v & 3)];
      accC = wmma_bf16(a, wc[kb], accC);
    }
    // ---- candidate, state update, output
#pragma unroll
    for (int v = 0; v < 8; ++v) {
      const int m = (lh << 3) + v;
      const int bb = b0 + m;
      const float* Xrow = X + (size_t)((bb * TDEC + t) * 3) * 256;
      float cand = tanhf(accC[v] + Xrow[512 + nb] + bC);
      float z = zt[v];
      float sOld = sF[m * 256 + nb];
      float st = (1.f - z) * cand + z * sOld;
      sF[m * 256 + nb] = st;
      sB[m * 256 + nb] = f32_to_bf16(st);
      out[(size_t)(bb * TDEC + t) * 256 + nb] = st;
    }
    __syncthreads();
  }
  // final state -> tail of output
  for (int i = threadIdx.x; i < 16 * 256; i += 512)
    out[(size_t)B_ * TDEC * 256 + (size_t)b0 * 256 + i] = sF[i];
}

// ==================================================================== launch
extern "C" void kernel_launch(void* const* d_in, const int* in_sizes, int n_in,
                              void* d_out, int out_size, void* d_ws, size_t ws_size,
                              hipStream_t stream) {
  (void)in_sizes; (void)n_in; (void)out_size; (void)ws_size;
  // setup_inputs() flat order: emb, all_h_enc, s_prev, then params in insertion
  // order (w,b pairs), then energy_projection_w.
  const float* emb    = (const float*)d_in[0];
  const float* henc   = (const float*)d_in[1];
  const float* sprev  = (const float*)d_in[2];
  const float* w_in_r = (const float*)d_in[3];  const float* b_in_r = (const float*)d_in[4];
  const float* w_in_z = (const float*)d_in[5];  const float* b_in_z = (const float*)d_in[6];
  const float* w_in_c = (const float*)d_in[7];  const float* b_in_c = (const float*)d_in[8];
  const float* w_h_r  = (const float*)d_in[9];  const float* b_h_r  = (const float*)d_in[10];
  const float* w_h_z  = (const float*)d_in[11]; const float* b_h_z  = (const float*)d_in[12];
  const float* w_h_c  = (const float*)d_in[13]; const float* b_h_c  = (const float*)d_in[14];
  const float* w_cx_r = (const float*)d_in[15]; const float* b_cx_r = (const float*)d_in[16];
  const float* w_cx_z = (const float*)d_in[17]; const float* b_cx_z = (const float*)d_in[18];
  const float* w_cx_c = (const float*)d_in[19]; const float* b_cx_c = (const float*)d_in[20];
  // d_in[21], d_in[22]: dec_projection_w/b — mathematically unused: the energy is
  // linear (no tanh), so the decoder term is constant per batch row and softmax
  // shift-invariance cancels it exactly.
  const float* w_enc  = (const float*)d_in[23]; const float* b_enc = (const float*)d_in[24];
  const float* v_en   = (const float*)d_in[25];

  char* p = (char*)d_ws;
  auto carve = [&](size_t bytes) -> void* {
    void* r = (void*)p;
    p += (bytes + 255) & ~(size_t)255;
    return r;
  };
  unsigned short* emb_bf  = (unsigned short*)carve(sizeof(unsigned short) * (size_t)B_ * TDEC * E_);
  unsigned short* win_bf  = (unsigned short*)carve(sizeof(unsigned short) * 3 * H_ * E_);
  unsigned short* whid_bf = (unsigned short*)carve(sizeof(unsigned short) * 3 * H_ * H_);
  unsigned short* wctx_bf = (unsigned short*)carve(sizeof(unsigned short) * 3 * H_ * H2_);
  float*          ct_f    = (float*)carve(sizeof(float) * (size_t)B_ * H2_);
  unsigned short* ct_bf   = (unsigned short*)carve(sizeof(unsigned short) * (size_t)B_ * H2_);
  float*          u_vec   = (float*)carve(sizeof(float) * (H2_ + 8));
  float*          c0      = u_vec + H2_;
  float*          Cgate   = (float*)carve(sizeof(float) * (size_t)B_ * 3 * H_);
  float*          X       = (float*)carve(sizeof(float) * (size_t)B_ * TDEC * 3 * H_);

  auto cvt = [&](const float* s, unsigned short* d, int n) {
    k_cvt_bf16<<<(n + 255) / 256, 256, 0, stream>>>(s, d, n);
  };
  cvt(emb,    emb_bf,               B_ * TDEC * E_);
  cvt(w_in_r, win_bf,               H_ * E_);
  cvt(w_in_z, win_bf + H_ * E_,     H_ * E_);
  cvt(w_in_c, win_bf + 2 * H_ * E_, H_ * E_);
  cvt(w_h_r,  whid_bf,               H_ * H_);
  cvt(w_h_z,  whid_bf + H_ * H_,     H_ * H_);
  cvt(w_h_c,  whid_bf + 2 * H_ * H_, H_ * H_);
  cvt(w_cx_r, wctx_bf,               H_ * H2_);
  cvt(w_cx_z, wctx_bf + H_ * H2_,    H_ * H2_);
  cvt(w_cx_c, wctx_bf + 2 * H_ * H2_,H_ * H2_);

  k_u_c0<<<1, 512, 0, stream>>>(w_enc, b_enc, v_en, u_vec, c0);
  k_attention<<<B_, 256, 0, stream>>>(henc, u_vec, c0, ct_f, ct_bf);
  k_ctx_gate<<<dim3(B_ / 16, 4, 3), 128, 0, stream>>>(
      (const unsigned*)ct_bf, (const unsigned*)wctx_bf,
      b_cx_r, b_cx_z, b_cx_c, b_in_r, b_in_z, b_in_c, Cgate);
  k_emb_gate<<<dim3(B_ * TDEC / 16, 4, 3), 128, 0, stream>>>(
      (const unsigned*)emb_bf, (const unsigned*)win_bf, Cgate, X);
  k_recurrence<<<B_ / 16, 512, 0, stream>>>(
      sprev, X, (const unsigned*)whid_bf, b_h_r, b_h_z, b_h_c, (float*)d_out);
}
